// QuantumClassifier_64501818851438
// MI455X (gfx1250) — matrix operand order repaired and checked by reading, not compiled, not run on hardware
//
#include <hip/hip_runtime.h>

typedef _Float16 h16;
typedef __attribute__((ext_vector_type(2)))  __fp16   v2fp16;
typedef __attribute__((ext_vector_type(16))) _Float16 v16h;
typedef __attribute__((ext_vector_type(8)))  float    v8f;

#define NQ       16
#define NSTATE   (1 << NQ)      // 65536 amplitudes
#define NTHREADS 1024           // 32 wave32s -> one full WGP
#define APT      (NSTATE / NTHREADS)  // 64 amps per thread

union H2U { unsigned u; h16 h[2]; };
union PKU { v2fp16 v; unsigned u; };
union B8  { v16h v; unsigned u[8]; };

// Single-instruction pack: v_cvt_pk_rtz_f16_f32 (re -> [15:0], im -> [31:16])
static __device__ inline unsigned pack2(float re, float im) {
  PKU p; p.v = __builtin_amdgcn_cvt_pkrtz(re, im); return p.u;
}

// One block = one batch sample. State lives entirely in LDS as packed fp16
// (re in [15:0], im in [31:16]) -> 256 KB, fits CDNA5's 320 KB/WGP LDS.
extern "C" __global__ __launch_bounds__(NTHREADS, 1)
void qsim16(const float* __restrict__ x, const float* __restrict__ wt,
            float* __restrict__ out)
{
  extern __shared__ unsigned smem[];
  unsigned* S    = smem;            // 65536 u32: packed fp16 (re,im)
  unsigned* Ua   = S + NSTATE;      // 256: A' pack( U_re, -U_im) per (m,k)
  unsigned* Ub   = Ua + 256;        // 256: A'' pack( U_im,  U_re)
  unsigned* gcol = Ub + 256;        // 16 : GF(2) columns of CNOT-ring gather map
  unsigned* g64  = gcol + 16;       // 64 : gather map on low 6 index bits
  float*    lo   = (float*)(g64 + 64); // 256 x (re,im): product over qubits 0..7
  float*    hi   = lo + 512;           // 256 x (re,im): product over qubits 8..15
  float*    red  = hi + 512;           // 32 floats: per-wave partials

  const int t = threadIdx.x;
  const int b = blockIdx.x;
  const int wave = t >> 5, L = t & 31;
  const int n = L & 15, hl = L >> 4;

  // ---- AngleEmbedding: RX(x_w)|0> is a product state; build 2 prefix tables ----
  if (t < 512) {
    const int half = t >> 8;          // 0: qubits 0..7, 1: qubits 8..15
    const int idx  = t & 255;
    float re = 1.f, im = 0.f;
    #pragma unroll
    for (int j = 0; j < 8; ++j) {
      const float th = 0.5f * x[b * NQ + (half * 8 + j)];
      const float c = __cosf(th), s = __sinf(th);
      if ((idx >> j) & 1) {           // amp factor for |1>: (re,im) * (-i*s)
        const float nre = s * im, nim = -s * re; re = nre; im = nim;
      } else {                        // amp factor for |0>: cos
        re *= c; im *= c;
      }
    }
    float* dst = half ? hi : lo;
    dst[idx * 2 + 0] = re; dst[idx * 2 + 1] = im;
  }
  __syncthreads();

  #pragma unroll 4
  for (int e = 0; e < APT; ++e) {
    const int i = t * APT + e;
    const float lre = lo[(i & 255) * 2], lim = lo[(i & 255) * 2 + 1];
    const float hre = hi[(i >> 8) * 2],  him = hi[(i >> 8) * 2 + 1];
    S[i] = pack2(lre * hre - lim * him, lre * him + lim * hre);
  }
  __syncthreads();

  for (int l = 0; l < 2; ++l) {
    const int r = l + 1;              // PennyLane default ranges: (l % 15) + 1

    // ---- Rot layer: 4 groups of 4 wires; each group is a 16x16 complex
    //      unitary U = kron(Rot_{4g..4g+3}) applied via f16 WMMA ----
    for (int g = 0; g < 4; ++g) {
      if (t < 256) {                  // build U[m][k] and its interleaved A packs
        const int m = t >> 4, k = t & 15;
        float ure = 1.f, uim = 0.f;
        #pragma unroll
        for (int j = 0; j < 4; ++j) {
          const float* ww = wt + ((l * NQ + (4 * g + j)) * 3);
          const float phi = ww[0], th = ww[1], om = ww[2];
          const float ct = __cosf(0.5f * th), st = __sinf(0.5f * th);
          const int mb = (m >> j) & 1, kb = (k >> j) & 1;
          float ang, sc;
          if (!mb && !kb)     { ang = -0.5f * (phi + om); sc =  ct; }
          else if (!mb && kb) { ang =  0.5f * (phi - om); sc = -st; }
          else if (mb && !kb) { ang = -0.5f * (phi - om); sc =  st; }
          else                { ang =  0.5f * (phi + om); sc =  ct; }
          const float ere = sc * __cosf(ang), eim = sc * __sinf(ang);
          const float nre = ure * ere - uim * eim;
          const float nim = ure * eim + uim * ere;
          ure = nre; uim = nim;
        }
        Ua[t] = pack2(ure, -uim);     // K-pair (2k: *S_re, 2k+1: *S_im) -> D_re
        Ub[t] = pack2(uim,  ure);     //                                 -> D_im
      }
      __syncthreads();

      // A operand (16x32 f16): lane holds row m=L%16; lanes>=16 hold K=16..31
      B8 a1, a2;
      #pragma unroll
      for (int v = 0; v < 8; ++v) {
        const int kk = hl * 8 + v;
        a1.u[v] = Ua[n * 16 + kk];
        a2.u[v] = Ub[n * 16 + kk];
      }

      const int sh = 4 * g;
      const unsigned lowm = (1u << sh) - 1u;
      const unsigned kofs = (unsigned)(hl * 8) << sh;

      // 4096 outer columns per sample -> 256 tiles of N=16; 32 waves handle
      // exactly 8 tiles each (compile-time trip count). Fully unrolled with
      // ping-pong B buffers: tile it+1's 8 LDS loads are issued before tile
      // it's 8 stores, so the WMMA wait covers loads only while stores drain.
      unsigned adr[2][8];
      B8 bbuf[2];
      {
        const int o = wave * 16 + n;                    // tile 0 for this wave
        const unsigned a0 = (((unsigned)o & lowm) |
                             (((unsigned)o >> sh) << (sh + 4))) | kofs;
        #pragma unroll
        for (int j = 0; j < 8; ++j) {
          adr[0][j] = a0 + ((unsigned)j << sh);
          bbuf[0].u[j] = S[adr[0][j]];
        }
      }
      #pragma unroll
      for (int it = 0; it < 8; ++it) {
        const int cur = it & 1, nxt = cur ^ 1;
        v8f cre = {0.f,0.f,0.f,0.f,0.f,0.f,0.f,0.f};
        v8f cim = {0.f,0.f,0.f,0.f,0.f,0.f,0.f,0.f};
        cre = __builtin_amdgcn_wmma_f32_16x16x32_f16(false, a1.v, false, bbuf[cur].v,
                                                     (short)0, cre, false, false);
        cim = __builtin_amdgcn_wmma_f32_16x16x32_f16(false, a2.v, false, bbuf[cur].v,
                                                     (short)0, cim, false, false);
        if (it < 7) {                                   // compile-time predicate
          const int o = (wave + (it + 1) * 32) * 16 + n;
          const unsigned a0 = (((unsigned)o & lowm) |
                               (((unsigned)o >> sh) << (sh + 4))) | kofs;
          #pragma unroll
          for (int j = 0; j < 8; ++j) {
            adr[nxt][j] = a0 + ((unsigned)j << sh);
            bbuf[nxt].u[j] = S[adr[nxt][j]];
          }
        }
        // store rows m = hl*8+d -> exactly the addresses this tile loaded from
        #pragma unroll
        for (int d = 0; d < 8; ++d)
          S[adr[cur][d]] = pack2(cre[d], cim[d]);
      }
      __syncthreads();
    }

    // ---- CNOT ring (w -> (w+r)%16, w=0..15) as a single GF(2)-linear gather:
    //      new[i] = old[G(i)],  G = f_0 . f_1 . ... . f_15 (each f self-inverse) ----
    if (t == 0) {
      for (int bb = 0; bb < NQ; ++bb) {
        unsigned v = 1u << bb;
        for (int w = NQ - 1; w >= 0; --w) {
          const int tq = (w + r) & (NQ - 1);
          v ^= ((v >> w) & 1u) << tq;                   // apply f_w
        }
        gcol[bb] = v;
      }
    }
    __syncthreads();
    if (t < 64) {
      unsigned v = 0;
      for (int bb = 0; bb < 6; ++bb) if ((t >> bb) & 1) v ^= gcol[bb];
      g64[t] = v;
    }
    __syncthreads();
    unsigned base = 0;
    #pragma unroll
    for (int bb = 6; bb < NQ; ++bb) if ((t >> (bb - 6)) & 1) base ^= gcol[bb];
    unsigned regs[APT];
    #pragma unroll
    for (int e = 0; e < APT; ++e) regs[e] = S[base ^ g64[e]];
    __syncthreads();                                    // all reads before any write
    #pragma unroll
    for (int e = 0; e < APT; ++e) S[t * APT + e] = regs[e];
    __syncthreads();
  }

  // ---- <Z_0>: qubit 0 <-> index bit 0 ----
  float acc = 0.f;
  #pragma unroll 4
  for (int e = 0; e < APT; ++e) {
    const int i = t * APT + e;
    H2U p; p.u = S[i];
    const float re = (float)p.h[0], im = (float)p.h[1];
    const float pr = re * re + im * im;
    acc += (i & 1) ? -pr : pr;
  }
  // wave32 shuffle reduction, then one 32-entry LDS stage
  #pragma unroll
  for (int off = 16; off > 0; off >>= 1) acc += __shfl_down(acc, off, 32);
  if (L == 0) red[wave] = acc;
  __syncthreads();
  if (t < 32) {
    float v = red[t];
    #pragma unroll
    for (int off = 16; off > 0; off >>= 1) v += __shfl_down(v, off, 32);
    if (t == 0) out[b] = v;
  }
}

extern "C" void kernel_launch(void* const* d_in, const int* in_sizes, int n_in,
                              void* d_out, int out_size, void* d_ws, size_t ws_size,
                              hipStream_t stream) {
  (void)in_sizes; (void)n_in; (void)d_ws; (void)ws_size;
  const float* x  = (const float*)d_in[0];   // (256, 16) f32
  const float* wt = (const float*)d_in[1];   // (2, 16, 3) f32
  float* out = (float*)d_out;                // (256,) f32
  const size_t smem = (size_t)(NSTATE + 256 + 256 + 16 + 64) * sizeof(unsigned)
                    + (size_t)(512 + 512 + 32) * sizeof(float); // 268736 B < 320 KB
  qsim16<<<out_size, NTHREADS, smem, stream>>>(x, wt, out);
}
